// Head_30279519437188
// MI455X (gfx1250) — matrix operand order repaired and checked by reading, not compiled, not run on hardware
//
#include <hip/hip_runtime.h>
#include <hip/hip_bf16.h>

// Fused causal attention head for MI455X (gfx1250), wave32 + WMMA f16.
// B=1024 batches, T=256, E=384, H=64, QKV=192. One workgroup per batch.

#define TSEQ 256
#define NEMBD 384
#define HSZ 64
#define NQKV 192
#define EP 392   // padded row length of W^T in LDS (f16 elems)
#define HP 72    // padded row length of q/k in LDS
#define TP 264   // padded row length of v^T in LDS

typedef _Float16 v8h  __attribute__((ext_vector_type(8)));
typedef _Float16 v16h __attribute__((ext_vector_type(16)));
typedef float    v8f  __attribute__((ext_vector_type(8)));

__device__ __forceinline__ v16h cat16(v8h lo, v8h hi) {
    v16h r;
#pragma unroll
    for (int i = 0; i < 8; ++i) { r[i] = lo[i]; r[i + 8] = hi[i]; }
    return r;
}

// two 8-f16 runs (p[0..7], p[16..23]) from LDS as two 16B loads -> one B-fragment register set
__device__ __forceinline__ v16h lds_frag32(const _Float16* p) {
    return cat16(*(const v8h*)p, *(const v8h*)(p + 16));
}

__global__ __launch_bounds__(256, 1) void attn_head_kernel(
    const float* __restrict__ x_all,   // [B, 256, 384]
    const float* __restrict__ Wqkv,    // [384, 192]
    float* __restrict__ out)           // [B, 256, 64]
{
    __shared__ __align__(16) _Float16 sWt[NQKV * EP];  // W^T: [n=192][k=384] f16
    __shared__ __align__(16) _Float16 sQ [TSEQ * HP];  // q * 1/8, row-major [t][d]
    __shared__ __align__(16) _Float16 sK [TSEQ * HP];  // k, row-major [t][d]
    __shared__ __align__(16) _Float16 sVT[HSZ  * TP];  // v^T: [d][t]
    __shared__ __align__(16) _Float16 sA [8 * 16 * 32];// per-wave P staging (C->A layout)

    const int tid  = threadIdx.x;
    const int wave = tid >> 5;
    const int lane = tid & 31;
    const int lh   = lane >> 4;   // which 16-lane half
    const int l16  = lane & 15;

    const float* x    = x_all + (size_t)blockIdx.x * (TSEQ * NEMBD);
    float*       outp = out   + (size_t)blockIdx.x * (TSEQ * HSZ);

    // ---------- stage W^T (f32 -> f16), coalesced global reads ----------
    for (int idx = tid; idx < NEMBD * NQKV; idx += 256) {
        const int kk = idx / NQKV;
        const int n  = idx - kk * NQKV;
        sWt[n * EP + kk] = (_Float16)Wqkv[idx];
    }
    __syncthreads();

    // ---------- phase 1: qkv = x @ W, x read from HBM exactly once ----------
#pragma unroll 1
    for (int half = 0; half < 2; ++half) {
        const int mt = wave + 8 * half;  // M tile 0..15
        v8f acc[12];
#pragma unroll
        for (int n = 0; n < 12; ++n)
#pragma unroll
            for (int r = 0; r < 8; ++r) acc[n][r] = 0.0f;

        const float* xrow = x + (mt * 16 + l16) * NEMBD + lh * 8;
        for (int ks = 0; ks < 12; ++ks) {           // K = 384 in steps of 32
            const float* xp = xrow + ks * 32;
            float4 f0 = *(const float4*)(xp);
            float4 f1 = *(const float4*)(xp + 4);
            float4 f2 = *(const float4*)(xp + 16);
            float4 f3 = *(const float4*)(xp + 20);
            v16h a;
            a[0]=(_Float16)f0.x; a[1]=(_Float16)f0.y; a[2]=(_Float16)f0.z; a[3]=(_Float16)f0.w;
            a[4]=(_Float16)f1.x; a[5]=(_Float16)f1.y; a[6]=(_Float16)f1.z; a[7]=(_Float16)f1.w;
            a[8]=(_Float16)f2.x; a[9]=(_Float16)f2.y; a[10]=(_Float16)f2.z; a[11]=(_Float16)f2.w;
            a[12]=(_Float16)f3.x; a[13]=(_Float16)f3.y; a[14]=(_Float16)f3.z; a[15]=(_Float16)f3.w;
#pragma unroll
            for (int n = 0; n < 12; ++n) {
                v16h bf = lds_frag32(&sWt[(n * 16 + l16) * EP + ks * 32 + lh * 8]);
                acc[n] = __builtin_amdgcn_wmma_f32_16x16x32_f16(
                    false, a, false, bf, (short)0, acc[n], false, false);
            }
        }
        // scatter accumulators: q (pre-scaled), k row-major; v transposed (packed b128)
        const int trow = mt * 16 + 8 * lh;
#pragma unroll
        for (int n = 0; n < 4; ++n) {
            const int col = n * 16 + l16;
#pragma unroll
            for (int r = 0; r < 8; ++r)
                sQ[(trow + r) * HP + col] = (_Float16)(acc[n][r] * 0.125f);
        }
#pragma unroll
        for (int n = 0; n < 4; ++n) {
            const int col = n * 16 + l16;
#pragma unroll
            for (int r = 0; r < 8; ++r)
                sK[(trow + r) * HP + col] = (_Float16)acc[4 + n][r];
        }
#pragma unroll
        for (int n = 0; n < 4; ++n) {
            const int d = n * 16 + l16;
            v8h pk;
#pragma unroll
            for (int r = 0; r < 8; ++r) pk[r] = (_Float16)acc[8 + n][r];
            *(v8h*)&sVT[d * TP + trow] = pk;   // 8 consecutive t -> one ds_store_b128
        }
    }
    __syncthreads();

    // ---------- phase 2+3: flash-style causal softmax(q k^T) v ----------
#pragma unroll 1
    for (int pass = 0; pass < 2; ++pass) {
        const int mt = pass ? (15 - wave) : wave;   // pair w with 15-w: uniform causal work
        // q A-fragments for this row block (constant across s)
        const _Float16* qb = &sQ[(mt * 16 + l16) * HP + lh * 8];
        const v16h aq0 = cat16(*(const v8h*)(qb),      *(const v8h*)(qb + 16));
        const v16h aq1 = cat16(*(const v8h*)(qb + 32), *(const v8h*)(qb + 48));

        v8f o[4];
        float mrow[8], lrow[8], corr[8];
#pragma unroll
        for (int n = 0; n < 4; ++n)
#pragma unroll
            for (int r = 0; r < 8; ++r) o[n][r] = 0.0f;
#pragma unroll
        for (int r = 0; r < 8; ++r) { mrow[r] = -1e30f; lrow[r] = 0.0f; }

        const int tg = mt * 16 + 8 * lh;            // global row of this lane's vgpr0

        for (int sp = 0; sp <= (mt >> 1); ++sp) {   // two s-tiles (32 keys) per step
            v8f s0, s1;
#pragma unroll
            for (int r = 0; r < 8; ++r) { s0[r] = 0.0f; s1[r] = 0.0f; }

            const _Float16* k0 = &sK[((2 * sp)     * 16 + l16) * HP + lh * 8];
            const _Float16* k1 = &sK[((2 * sp + 1) * 16 + l16) * HP + lh * 8];
            s0 = __builtin_amdgcn_wmma_f32_16x16x32_f16(false, aq0, false, lds_frag32(k0),      (short)0, s0, false, false);
            s1 = __builtin_amdgcn_wmma_f32_16x16x32_f16(false, aq0, false, lds_frag32(k1),      (short)0, s1, false, false);
            s0 = __builtin_amdgcn_wmma_f32_16x16x32_f16(false, aq1, false, lds_frag32(k0 + 32), (short)0, s0, false, false);
            s1 = __builtin_amdgcn_wmma_f32_16x16x32_f16(false, aq1, false, lds_frag32(k1 + 32), (short)0, s1, false, false);

            const int s0g = sp * 32 + l16;          // global key index for s0 column
#pragma unroll
            for (int r = 0; r < 8; ++r) {
                const int t = tg + r;
                float v0 = (s0g      <= t) ? s0[r] : -1e30f;  // causal mask
                float v1 = (s0g + 16 <= t) ? s1[r] : -1e30f;
                float tm = fmaxf(v0, v1);
                tm = fmaxf(tm, __shfl_xor(tm, 1));
                tm = fmaxf(tm, __shfl_xor(tm, 2));
                tm = fmaxf(tm, __shfl_xor(tm, 4));
                tm = fmaxf(tm, __shfl_xor(tm, 8));
                const float nm = fmaxf(mrow[r], tm);
                const float c  = __expf(mrow[r] - nm);
                corr[r] = c; mrow[r] = nm;
                const float p0 = __expf(v0 - nm);
                const float p1 = __expf(v1 - nm);
                float ps = p0 + p1;
                ps += __shfl_xor(ps, 1);
                ps += __shfl_xor(ps, 2);
                ps += __shfl_xor(ps, 4);
                ps += __shfl_xor(ps, 8);
                lrow[r] = lrow[r] * c + ps;
                // stage P in per-wave buffer, row = t_local, col = s_local (C layout out)
                sA[(wave * 16 + 8 * lh + r) * 32 + l16]      = (_Float16)p0;
                sA[(wave * 16 + 8 * lh + r) * 32 + 16 + l16] = (_Float16)p1;
            }
#pragma unroll
            for (int n = 0; n < 4; ++n)
#pragma unroll
                for (int r = 0; r < 8; ++r) o[n][r] *= corr[r];

            asm volatile("s_wait_dscnt 0" ::: "memory");  // P staging visible wave-wide

            // reload P as an A-fragment (16x32 f16 layout)
            const _Float16* ab = &sA[(wave * 16 + l16) * 32 + lh * 8];
            const v16h ap = cat16(*(const v8h*)ab, *(const v8h*)(ab + 16));
#pragma unroll
            for (int n = 0; n < 4; ++n) {
                const _Float16* vb = &sVT[(n * 16 + l16) * TP + sp * 32 + lh * 8];
                o[n] = __builtin_amdgcn_wmma_f32_16x16x32_f16(
                    false, ap, false, lds_frag32(vb), (short)0, o[n], false, false);
            }
        }

        // epilogue: normalize and store (coalesced across lanes per vgpr row)
#pragma unroll
        for (int n = 0; n < 4; ++n) {
#pragma unroll
            for (int r = 0; r < 8; ++r) {
                const int t = tg + r;
                outp[(size_t)t * HSZ + n * 16 + l16] = o[n][r] / lrow[r];
            }
        }
    }
}

extern "C" void kernel_launch(void* const* d_in, const int* in_sizes, int n_in,
                              void* d_out, int out_size, void* d_ws, size_t ws_size,
                              hipStream_t stream) {
    (void)n_in; (void)d_ws; (void)ws_size; (void)out_size;
    const float* x = (const float*)d_in[0];   // [B,256,384] f32
    const float* W = (const float*)d_in[1];   // [384,192]  f32
    float* out     = (float*)d_out;           // [B,256,64] f32
    const int batches = in_sizes[0] / (TSEQ * NEMBD);
    attn_head_kernel<<<batches, 256, 0, stream>>>(x, W, out);
}
